// LSTMActor_final_69020124447515
// MI455X (gfx1250) — matrix operand order repaired and checked by reading, not compiled, not run on hardware
//
#include <hip/hip_runtime.h>
#include <hip/hip_bf16.h>

typedef __attribute__((ext_vector_type(16))) _Float16 v16h;
typedef __attribute__((ext_vector_type(8)))  _Float16 v8h;
typedef __attribute__((ext_vector_type(8)))  float    v8f;

#define B_    512
#define T_    64
#define H_    15
#define NROWS (B_*T_)     // 32768
#define STEPS (T_-1)      // 63

// workspace byte offsets (all 256B aligned)
#define COMB_OFF   0          // 32x64  f16 swizzled (4 B-frags)   : 4096 B
#define W1P_OFF    4096       // 32x256 f16 swizzled (16 frags)    : 16384 B
#define W2P_OFF    20480      // 256x256 f16 swizzled (128 frags)  : 131072 B
#define BIASC_OFF  151552     // 64 f32 (bias_xh+bias_hh padded)   : 256 B
#define FEAT_OFF   151808     // 32768 x 32 f16 row-major          : 2 MB

__device__ __forceinline__ v16h cat8(v8h lo, v8h hi) {
  return __builtin_shufflevector(lo, hi, 0,1,2,3,4,5,6,7,8,9,10,11,12,13,14,15);
}
__device__ __forceinline__ v8f wmma16(v16h a, v16h b, v8f c) {
  return __builtin_amdgcn_wmma_f32_16x16x32_f16(false, a, false, b, (short)0, c, false, false);
}
// fast transcendentals: v_exp_f32 / v_rcp_f32 / v_rsq_f32, no IEEE div chains
__device__ __forceinline__ float fast_sigmoid(float x) {
  return __builtin_amdgcn_rcpf(1.0f + __expf(-x));
}
__device__ __forceinline__ float fast_tanh(float x) {
  return 1.0f - 2.0f * __builtin_amdgcn_rcpf(1.0f + __expf(2.0f * x));
}

// deterministic stand-in for jax.random.normal(key 42): splitmix64 + Box-Muller
__device__ __forceinline__ float hash_norm(unsigned idx) {
  unsigned long long x = (unsigned long long)idx * 0x9E3779B97F4A7C15ULL + 0xD1B54A32D192ED03ULL;
  x ^= x >> 30; x *= 0xBF58476D1CE4E5B9ULL;
  x ^= x >> 27; x *= 0x94D049BB133111EBULL;
  x ^= x >> 31;
  float u1 = ((float)(unsigned)(x >> 40) + 1.0f) * (1.0f / 16777217.0f);
  float u2 = ((float)(unsigned)(x & 0xFFFFFFULL)) * (1.0f / 16777216.0f);
  return sqrtf(-2.0f * __logf(u1)) * __cosf(6.28318530717958647f * u2);
}

// ---------------------------------------------------------------------------
// Kernel 0: convert weights to f16 in WMMA B-fragment-swizzled order.
// B-fragment (32x16 tile): lane L holds col n = nt*16 + (L&15),
// K = kt*32 + (L/16)*16 + j  (j = 0..15), stored contiguous per lane.
// ---------------------------------------------------------------------------
__global__ void prep_kernel(const float* __restrict__ wxh, const float* __restrict__ whh,
                            const float* __restrict__ bxh, const float* __restrict__ bhh,
                            const float* __restrict__ w1,  const float* __restrict__ w2,
                            char* __restrict__ ws) {
  int idx = blockIdx.x * 256 + threadIdx.x;
  _Float16* comb = (_Float16*)(ws + COMB_OFF);
  _Float16* w1p  = (_Float16*)(ws + W1P_OFF);
  _Float16* w2p  = (_Float16*)(ws + W2P_OFF);
  float*    bc   = (float*)   (ws + BIASC_OFF);
  if (idx < 2048) {                       // comb: KT=1, NT=4
    int o = idx, frag = o >> 9, lane = (o >> 4) & 31, j = o & 15;
    int n = frag * 16 + (lane & 15);
    int K = ((lane >> 4) << 4) + j;
    float v = 0.0f;
    if (n < 60) {
      if (K < 5)       v = wxh[K * 60 + n];
      else if (K < 20) v = whh[(K - 5) * 60 + n];
    }
    comb[o] = (_Float16)v;
  } else if (idx < 2048 + 8192) {         // w1p: KT=1, NT=16
    int o = idx - 2048, frag = o >> 9, lane = (o >> 4) & 31, j = o & 15;
    int n = (frag & 15) * 16 + (lane & 15);
    int K = ((lane >> 4) << 4) + j;
    w1p[o] = (_Float16)((K < 23) ? w1[K * 256 + n] : 0.0f);
  } else if (idx < 2048 + 8192 + 65536) { // w2p: KT=8, NT=16
    int o = idx - (2048 + 8192), frag = o >> 9, lane = (o >> 4) & 31, j = o & 15;
    int kt = frag >> 4, nt = frag & 15;
    int n = nt * 16 + (lane & 15);
    int K = kt * 32 + ((lane >> 4) << 4) + j;
    w2p[o] = (_Float16)w2[K * 256 + n];
  } else if (idx < 2048 + 8192 + 65536 + 64) {
    int n = idx - (2048 + 8192 + 65536);
    bc[n] = (n < 60) ? (bxh[n] + bhh[n]) : 0.0f;
  }
}

// ---------------------------------------------------------------------------
// Kernel 1: geometric prep + 63-step LSTM, fused.
// 256 blocks x 256 threads (8 waves). Block owns 2 batch elems = 128 rows;
// each wave owns a 16-row tile, 4x v_wmma_f32_16x16x32_f16 per step.
// State slice staged via global_load_async_to_lds_b128 (ASYNCcnt DMA path).
// ---------------------------------------------------------------------------
__global__ __launch_bounds__(256)
void lstm_kernel(const float* __restrict__ state, char* __restrict__ ws) {
  __shared__ __align__(16) float st[2][64][8];
  __shared__ float csa[2][64][2];
  __shared__ float biasc[64];
  __shared__ __align__(16) _Float16 atile[8][16][32];   // per-wave A (x|h) f16
  __shared__ float gates[8][16][64];                    // per-wave gate outputs
  __shared__ float cbuf[8][16][16];                     // per-wave cell state

  const int tid  = threadIdx.x;
  const int lane = tid & 31;
  const int w    = tid >> 5;
  const int g    = lane >> 4;
  const int ln   = lane & 15;
  const int bid  = blockIdx.x;

  // stage the block's 2 batch elements of state (4KB) via async DMA to LDS:
  // one 16B async load per thread, tracked with ASYNCcnt.
  {
    unsigned      ldsoff = (unsigned)(unsigned long long)(void*)st + tid * 16u;
    unsigned long long ga = (unsigned long long)(const void*)(state + (size_t)bid * 128 * 8)
                          + (unsigned long long)tid * 16ull;
    asm volatile("global_load_async_to_lds_b128 %0, %1, off"
                 :: "v"(ldsoff), "v"(ga) : "memory");
  }
  if (tid < 64) biasc[tid] = ((const float*)(ws + BIASC_OFF))[tid];
  for (int i = tid; i < 8 * 16 * 32; i += 256) ((_Float16*)atile)[i] = (_Float16)0.0f;
  for (int i = tid; i < 8 * 16 * 16; i += 256) ((float*)cbuf)[i] = 0.0f;
  asm volatile("s_wait_asynccnt 0x0" ::: "memory");
  __syncthreads();
  for (int i = tid; i < 128; i += 256) {
    int bl2 = i >> 6, j = i & 63;
    float ang = 1.57079632679489662f - st[bl2][j][7];
    csa[bl2][j][0] = __cosf(ang);
    csa[bl2][j][1] = __sinf(ang);
  }
  // resident gate-weight B fragments (32 VGPRs) + per-lane gate bias splats
  const v8h* combp = (const v8h*)(ws + COMB_OFF);
  v16h Bw[4];
#pragma unroll
  for (int nt = 0; nt < 4; nt++) {
    int hbase = nt * 512 + lane * 16;
    Bw[nt] = cat8(combp[hbase >> 3], combp[(hbase >> 3) + 1]);
  }
  __syncthreads();
  float bgate[4];
#pragma unroll
  for (int nt = 0; nt < 4; nt++) bgate[nt] = biasc[nt * 16 + ln];

  const int bl    = w >> 2;          // batch-local index for this wave
  const int tbase = (w & 3) * 16;    // t offset of this wave's rows

  for (int s = 0; s < STEPS; s++) {
    // (1) step-s geometric features -> A cols 0..4 (lanes 0..15, one row each)
    if (lane < 16) {
      int m = lane, t = tbase + m;
      int j = s + (s >= t ? 1 : 0);
      float dx  = st[bl][j][0] - st[bl][t][0];
      float dy  = st[bl][j][1] - st[bl][t][1];
      float dvx = st[bl][j][2] - st[bl][t][2];
      float dvy = st[bl][j][3] - st[bl][t][3];
      float ca = csa[bl][j][0], sa = csa[bl][j][1];
      float xp = dx * ca - dy * sa;
      float yp = dx * sa + dy * ca;
      float xv = dvx * ca - dvy * sa;
      float yv = dvx * sa + dvy * ca;
      float d2  = xp * xp + yp * yp;
      float inv = __builtin_amdgcn_rsqf(d2);   // 1/r
      float rr  = d2 * inv;                    // r
      float rt  = fast_sigmoid(1.0f - 5.0f * (rr - 0.2f));
      atile[w][m][0] = (_Float16)(xp * inv);
      atile[w][m][1] = (_Float16)(yp * inv);
      atile[w][m][2] = (_Float16)xv;
      atile[w][m][3] = (_Float16)yv;
      atile[w][m][4] = (_Float16)rt;
    }
    __syncthreads();
    // (2) A fragment (ds_load_b128 x2); 4 WMMAs with bias pre-loaded into C
    v16h A = cat8(*(const v8h*)&atile[w][ln][g * 16],
                  *(const v8h*)&atile[w][ln][g * 16 + 8]);
#pragma unroll
    for (int nt = 0; nt < 4; nt++) {
      float bb = bgate[nt];
      v8f cb = {bb, bb, bb, bb, bb, bb, bb, bb};
      v8f d = wmma16(A, Bw[nt], cb);
#pragma unroll
      for (int v = 0; v < 8; v++) gates[w][g * 8 + v][nt * 16 + ln] = d[v];
    }
    __syncthreads();
    // (3) elementwise LSTM cell update: 240 elems / wave, predicated unroll
#pragma unroll
    for (int k = 0; k < 8; k++) {
      int e = lane + k * 32;
      if (e < 16 * H_) {
        int m = e / H_, u = e % H_;
        float gi = fast_sigmoid(gates[w][m][u]);
        float gf = fast_sigmoid(gates[w][m][H_ + u]);
        float gg = fast_tanh(   gates[w][m][2 * H_ + u]);
        float go = fast_sigmoid(gates[w][m][3 * H_ + u]);
        float c = gf * cbuf[w][m][u] + gi * gg;
        cbuf[w][m][u] = c;
        atile[w][m][5 + u] = (_Float16)(go * fast_tanh(c));  // h -> A cols 5..19
      }
    }
    __syncthreads();
  }

  // write feat rows [h(15)|state(8)|0...] as f16
  if (lane < 16) {
    int m = lane, t = tbase + m;
    int r = bid * 128 + w * 16 + m;
    _Float16* fout = (_Float16*)(ws + FEAT_OFF) + (size_t)r * 32;
    __align__(16) _Float16 tmp[32];
#pragma unroll
    for (int cI = 0; cI < 32; cI++) {
      float v;
      if (cI < 15)      v = (float)atile[w][m][5 + cI];
      else if (cI < 23) v = st[bl][t][cI - 15];
      else              v = 0.0f;
      tmp[cI] = (_Float16)v;
    }
#pragma unroll
    for (int ch = 0; ch < 4; ch++) ((v8h*)fout)[ch] = *(v8h*)&tmp[ch * 8];
  }
}

// ---------------------------------------------------------------------------
// Kernel 2: MLP 23->256->256 + fused heads + tanh-gaussian sampling.
// 512 blocks x 128 threads (4 waves); wave = one 16-row tile.
// Layer2 chains 8 accumulating WMMAs per N-tile; heads fused into epilogue.
// ---------------------------------------------------------------------------
__global__ __launch_bounds__(128)
void mlp_kernel(const char* __restrict__ ws,
                const float* __restrict__ b1, const float* __restrict__ b2,
                const float* __restrict__ wmu, const float* __restrict__ bmu,
                const float* __restrict__ wls, const float* __restrict__ bls,
                float* __restrict__ out) {
  __shared__ __align__(16) _Float16 f1buf[4][16][256];
  __shared__ float headacc[4][16][4];

  const int tid  = threadIdx.x;
  const int lane = tid & 31;
  const int w    = tid >> 5;
  const int g    = lane >> 4;
  const int ln   = lane & 15;
  const int rowbase = (blockIdx.x * 4 + w) * 16;

  for (int i = tid; i < 4 * 16 * 4; i += 128) ((float*)headacc)[i] = 0.0f;

  // A0 fragment straight from global feat rows
  const _Float16* arow = (const _Float16*)(ws + FEAT_OFF) + (size_t)(rowbase + ln) * 32;
  v16h A0 = cat8(*(const v8h*)(arow + g * 16), *(const v8h*)(arow + g * 16 + 8));

  // layer 1: 16 WMMAs with bias in C, relu, stage f16 activations to LDS
  const v8h* w1p = (const v8h*)(ws + W1P_OFF);
#pragma unroll
  for (int nt = 0; nt < 16; nt++) {
    int hbase = nt * 512 + lane * 16;
    v16h Bf = cat8(w1p[hbase >> 3], w1p[(hbase >> 3) + 1]);
    int n = nt * 16 + ln;
    float bb = b1[n];
    v8f cb = {bb, bb, bb, bb, bb, bb, bb, bb};
    v8f d = wmma16(A0, Bf, cb);
#pragma unroll
    for (int v = 0; v < 8; v++) {
      float val = d[v];
      f1buf[w][g * 8 + v][n] = (_Float16)(val > 0.0f ? val : 0.0f);
    }
  }
  __syncthreads();

  // layer 2 (256x256, K-chained WMMA accumulation) + fused head dot products
  const v8h* w2p = (const v8h*)(ws + W2P_OFF);
  float ha[8][4];
#pragma unroll
  for (int v = 0; v < 8; v++)
#pragma unroll
    for (int c = 0; c < 4; c++) ha[v][c] = 0.0f;

  for (int nt = 0; nt < 16; nt++) {
    if (nt < 15)  // pull next weight column-block toward L2/L0 early
      __builtin_prefetch((const void*)&w2p[(((nt + 1) * 512) + lane * 16) >> 3], 0, 0);
    int n = nt * 16 + ln;
    float bb = b2[n];
    v8f acc = {bb, bb, bb, bb, bb, bb, bb, bb};
#pragma unroll
    for (int kt = 0; kt < 8; kt++) {
      v16h Af = cat8(*(const v8h*)&f1buf[w][ln][kt * 32 + g * 16],
                     *(const v8h*)&f1buf[w][ln][kt * 32 + g * 16 + 8]);
      int hbase = (kt * 16 + nt) * 512 + lane * 16;
      v16h Bf = cat8(w2p[hbase >> 3], w2p[(hbase >> 3) + 1]);
      acc = wmma16(Af, Bf, acc);
    }
    float wm0 = wmu[2 * n], wm1 = wmu[2 * n + 1];
    float wl0 = wls[2 * n], wl1 = wls[2 * n + 1];
#pragma unroll
    for (int v = 0; v < 8; v++) {
      float f2 = acc[v];
      f2 = f2 > 0.0f ? f2 : 0.0f;
      ha[v][0] += f2 * wm0; ha[v][1] += f2 * wm1;
      ha[v][2] += f2 * wl0; ha[v][3] += f2 * wl1;
    }
  }
#pragma unroll
  for (int v = 0; v < 8; v++) {
    int m = g * 8 + v;
    atomicAdd(&headacc[w][m][0], ha[v][0]);
    atomicAdd(&headacc[w][m][1], ha[v][1]);
    atomicAdd(&headacc[w][m][2], ha[v][2]);
    atomicAdd(&headacc[w][m][3], ha[v][3]);
  }
  __syncthreads();

  if (lane < 16) {
    int r = rowbase + lane;
    float lp = 0.0f, act[2];
#pragma unroll
    for (int c = 0; c < 2; c++) {
      float mu = fast_tanh(headacc[w][lane][c]     + bmu[c]);
      float ls = fast_tanh(headacc[w][lane][2 + c] + bls[c]);
      float log_std = -20.0f + 11.0f * (ls + 1.0f);
      float sd = __expf(log_std);
      float e  = hash_norm((unsigned)(r * 2 + c));
      float z  = mu + sd * e;
      float a  = fast_tanh(z);
      act[c] = a;
      lp += -0.5f * e * e - log_std - 0.91893853320467274f
            - __logf(1.0f - a * a + 1e-7f);
    }
    out[(size_t)r * 2 + 0] = act[0];
    out[(size_t)r * 2 + 1] = act[1];
    out[(size_t)NROWS * 2 + r] = lp;
  }
}

extern "C" void kernel_launch(void* const* d_in, const int* in_sizes, int n_in,
                              void* d_out, int out_size, void* d_ws, size_t ws_size,
                              hipStream_t stream) {
  (void)in_sizes; (void)n_in; (void)out_size; (void)ws_size;
  const float* state = (const float*)d_in[0];
  const float* wxh   = (const float*)d_in[1];
  const float* whh   = (const float*)d_in[2];
  const float* bxh   = (const float*)d_in[3];
  const float* bhh   = (const float*)d_in[4];
  const float* w1    = (const float*)d_in[5];
  const float* b1    = (const float*)d_in[6];
  const float* w2    = (const float*)d_in[7];
  const float* b2    = (const float*)d_in[8];
  const float* wmu   = (const float*)d_in[9];
  const float* bmu   = (const float*)d_in[10];
  const float* wls   = (const float*)d_in[11];
  const float* bls   = (const float*)d_in[12];
  char* ws = (char*)d_ws;

  prep_kernel<<<297, 256, 0, stream>>>(wxh, whh, bxh, bhh, w1, w2, ws);
  lstm_kernel<<<NROWS / 128, 256, 0, stream>>>(state, ws);
  mlp_kernel<<<NROWS / 64, 128, 0, stream>>>(ws, b1, b2, wmu, bmu, wls, bls, (float*)d_out);
}
